// Block_89970974917283
// MI455X (gfx1250) — matrix operand order repaired and checked by reading, not compiled, not run on hardware
//
#include <hip/hip_runtime.h>
#include <hip/hip_bf16.h>
#include <math.h>

// ---------------------------------------------------------------------------
// ConvNeXt block for MI455X (gfx1250, wave32, WMMA).
//   x:(16,512,32,32) f32 NCHW -> dw7x7 -> LN(C) -> 512->2048 GELU -> 2048->512
//   -> gamma -> +x
// GEMMs (68.7 GFLOP) run on v_wmma_f32_16x16x32_f16.
// A tiles are staged memory->LDS with gfx1250 async copies (ASYNCcnt),
// B tiles with coalesced b128 loads + transposed LDS stores.
// ---------------------------------------------------------------------------

#define DIM 512
#define HID 2048
#define NB  16
#define HW  32
#define NPOS (NB * HW * HW)          // 16384 spatial positions (GEMM M)

typedef __attribute__((ext_vector_type(16))) _Float16 v16h;
typedef __attribute__((ext_vector_type(8)))  float    v8f;

// ---------------- Kernel 1: depthwise 7x7 conv + LayerNorm -> f16 A ---------
__global__ __launch_bounds__(256)
void dwconv_ln_kernel(const float* __restrict__ x,
                      const float* __restrict__ dw_w,
                      const float* __restrict__ dw_b,
                      const float* __restrict__ ln_w,
                      const float* __restrict__ ln_b,
                      _Float16* __restrict__ Aout) {
    const int pos = blockIdx.x;                 // 0..16383  (NHWC row index)
    const int w   = pos & 31;
    const int h   = (pos >> 5) & 31;
    const int nb  = pos >> 10;
    const int t   = threadIdx.x;

    __shared__ float red[256];

    float acc[2];
    #pragma unroll
    for (int i = 0; i < 2; ++i) {
        const int c = t + i * 256;
        const float* xc = x + ((size_t)(nb * DIM + c) << 10);   // HW plane
        const float* wc = dw_w + c * 49;
        float s = dw_b[c];
        #pragma unroll
        for (int dh = 0; dh < 7; ++dh) {
            const int hh = h + dh - 3;
            if (hh < 0 || hh >= 32) continue;
            #pragma unroll
            for (int dw = 0; dw < 7; ++dw) {
                const int ww = w + dw - 3;
                if (ww < 0 || ww >= 32) continue;
                s += wc[dh * 7 + dw] * xc[(hh << 5) + ww];
            }
        }
        acc[i] = s;
    }

    // mean over 512 channels
    red[t] = acc[0] + acc[1];
    __syncthreads();
    for (int off = 128; off > 0; off >>= 1) {
        if (t < off) red[t] += red[t + off];
        __syncthreads();
    }
    const float mu = red[0] * (1.0f / 512.0f);
    __syncthreads();

    // variance
    const float d0 = acc[0] - mu, d1 = acc[1] - mu;
    red[t] = d0 * d0 + d1 * d1;
    __syncthreads();
    for (int off = 128; off > 0; off >>= 1) {
        if (t < off) red[t] += red[t + off];
        __syncthreads();
    }
    const float rstd = rsqrtf(red[0] * (1.0f / 512.0f) + 1e-6f);

    #pragma unroll
    for (int i = 0; i < 2; ++i) {
        const int c = t + i * 256;
        const float y = (acc[i] - mu) * rstd * ln_w[c] + ln_b[c];
        Aout[(size_t)pos * DIM + c] = (_Float16)y;
    }
}

// ---------------- Kernel 2: weight f32 -> f16 conversion --------------------
__global__ __launch_bounds__(256)
void cvt_weights_kernel(const float* __restrict__ w1,
                        const float* __restrict__ w2,
                        _Float16* __restrict__ w1h,
                        _Float16* __restrict__ w2h) {
    const int i = blockIdx.x * 256 + threadIdx.x;   // 0 .. 2*1048576-1
    const int n1 = DIM * HID;
    if (i < n1)            w1h[i]      = (_Float16)w1[i];
    else if (i < 2 * n1)   w2h[i - n1] = (_Float16)w2[i - n1];
}

// ---------------- WMMA fragment loaders (wave32, contiguous k order: k = 16*(lane/16) + s) --------
__device__ __forceinline__ v16h load_a_frag(const _Float16* __restrict__ As,
                                            int mbase, int lane) {
    // lane<16: row m+lane,    K = 0..15
    // lane>=16: row m+lane-16, K = 16..31
    const _Float16* row = As + (size_t)(mbase + (lane & 15)) * 32;
    const int ko = (lane >> 4) * 16;
    union { v16h v; float4 q[2]; } u;
    u.q[0] = *(const float4*)(row + ko);
    u.q[1] = *(const float4*)(row + ko + 8);
    return u.v;
}

__device__ __forceinline__ v16h load_b_frag(const _Float16* __restrict__ Bs,
                                            int nbase, int lane) {
    // Bs staged transposed: Bs[n][k].  lane<16: col n+lane, K=0..15;
    // lane>=16: col n+lane-16, K=16..31  -> 16 contiguous halfs.
    const _Float16* col = Bs + (size_t)(nbase + (lane & 15)) * 32 + (lane >> 4) * 16;
    union { v16h v; float4 q[2]; } u;
    u.q[0] = *(const float4*)(col);
    u.q[1] = *(const float4*)(col + 8);
    return u.v;
}

// gfx1250 async copy: 16 bytes global -> LDS per lane, tracked by ASYNCcnt.
__device__ __forceinline__ void async_copy_b128(const _Float16* gptr,
                                                _Float16* lds_ptr) {
    const unsigned lds_off = (unsigned)(uintptr_t)lds_ptr;  // low 32b = LDS addr
    asm volatile("global_load_async_to_lds_b128 %0, %1, off"
                 :: "v"(lds_off), "v"(gptr)
                 : "memory");
}
__device__ __forceinline__ void async_wait0() {
    asm volatile("s_wait_asynccnt 0" ::: "memory");
}

// ---------------- Kernel 3: GEMM1 (16384x512 @ 512x2048) + bias + GELU ------
__global__ __launch_bounds__(256)
void gemm1_kernel(const _Float16* __restrict__ A,   // [NPOS][DIM]
                  const _Float16* __restrict__ B,   // w1h [DIM][HID]
                  const float* __restrict__ bias,   // b1 [HID]
                  _Float16* __restrict__ H) {       // [NPOS][HID]
    __shared__ _Float16 As[128 * 32];
    __shared__ _Float16 Bs[128 * 32];               // transposed [n][k]

    const int nblk = blockIdx.x, mblk = blockIdx.y;
    const int t = threadIdx.x;
    const int wave = t >> 5, lane = t & 31;
    const int wm = wave >> 1, wn = wave & 1;        // 4x2 wave grid -> 32x64/wave

    v8f acc[2][4] = {};

    for (int k0 = 0; k0 < DIM; k0 += 32) {
        // A tile 128x32: async global->LDS, 16B per lane, two passes
        {
            const int rowp = t >> 2, colp = (t & 3) * 8;
            #pragma unroll
            for (int p = 0; p < 2; ++p) {
                const int r = rowp + p * 64;
                async_copy_b128(A + (size_t)(mblk * 128 + r) * DIM + k0 + colp,
                                &As[r * 32 + colp]);
            }
        }
        // B tile 32x128 -> Bs[n][k]; coalesced b128 reads along n
        {
            const int kk  = t >> 3;           // 0..31
            const int nn0 = (t & 7) * 16;     // 0..112
            union { float4 q[2]; _Float16 h[16]; } u;
            const _Float16* g = B + (size_t)(k0 + kk) * HID + nblk * 128 + nn0;
            u.q[0] = *(const float4*)(g);
            u.q[1] = *(const float4*)(g + 8);
            #pragma unroll
            for (int j = 0; j < 16; ++j)
                Bs[(nn0 + j) * 32 + kk] = u.h[j];
        }
        if (k0 + 32 < DIM)
            __builtin_prefetch(B + (size_t)(k0 + 32 + (t >> 3)) * HID + nblk * 128, 0, 1);
        async_wait0();
        __syncthreads();

        #pragma unroll
        for (int mi = 0; mi < 2; ++mi) {
            const v16h a = load_a_frag(As, wm * 32 + mi * 16, lane);
            #pragma unroll
            for (int ni = 0; ni < 4; ++ni) {
                const v16h b = load_b_frag(Bs, wn * 64 + ni * 16, lane);
                acc[mi][ni] = __builtin_amdgcn_wmma_f32_16x16x32_f16(
                    false, a, false, b, (short)0, acc[mi][ni], false, false);
            }
        }
        __syncthreads();
    }

    // epilogue: bias + exact GELU -> f16 hidden
    #pragma unroll
    for (int mi = 0; mi < 2; ++mi) {
        const int mbase = mblk * 128 + wm * 32 + mi * 16 + (lane >> 4) * 8;
        #pragma unroll
        for (int ni = 0; ni < 4; ++ni) {
            const int n = nblk * 128 + wn * 64 + ni * 16 + (lane & 15);
            const float bn = bias[n];
            #pragma unroll
            for (int g = 0; g < 8; ++g) {
                float v = acc[mi][ni][g] + bn;
                v = 0.5f * v * (1.0f + erff(v * 0.7071067811865475f));
                H[(size_t)(mbase + g) * HID + n] = (_Float16)v;
            }
        }
    }
}

// ---- Kernel 4: GEMM2 (16384x2048 @ 2048x512) + bias + gamma + residual ----
__global__ __launch_bounds__(256)
void gemm2_kernel(const _Float16* __restrict__ H,   // [NPOS][HID]
                  const _Float16* __restrict__ B,   // w2h [HID][DIM]
                  const float* __restrict__ bias,   // b2 [DIM]
                  const float* __restrict__ gamma,  // [DIM]
                  const float* __restrict__ x,      // NCHW residual
                  float* __restrict__ out) {        // NCHW
    __shared__ _Float16 As[128 * 32];
    __shared__ _Float16 Bs[128 * 32];

    const int nblk = blockIdx.x, mblk = blockIdx.y;
    const int t = threadIdx.x;
    const int wave = t >> 5, lane = t & 31;
    const int wm = wave >> 1, wn = wave & 1;

    v8f acc[2][4] = {};

    for (int k0 = 0; k0 < HID; k0 += 32) {
        {
            const int rowp = t >> 2, colp = (t & 3) * 8;
            #pragma unroll
            for (int p = 0; p < 2; ++p) {
                const int r = rowp + p * 64;
                async_copy_b128(H + (size_t)(mblk * 128 + r) * HID + k0 + colp,
                                &As[r * 32 + colp]);
            }
        }
        {
            const int kk  = t >> 3;
            const int nn0 = (t & 7) * 16;
            union { float4 q[2]; _Float16 h[16]; } u;
            const _Float16* g = B + (size_t)(k0 + kk) * DIM + nblk * 128 + nn0;
            u.q[0] = *(const float4*)(g);
            u.q[1] = *(const float4*)(g + 8);
            #pragma unroll
            for (int j = 0; j < 16; ++j)
                Bs[(nn0 + j) * 32 + kk] = u.h[j];
        }
        if (k0 + 32 < HID)
            __builtin_prefetch(B + (size_t)(k0 + 32 + (t >> 3)) * DIM + nblk * 128, 0, 1);
        async_wait0();
        __syncthreads();

        #pragma unroll
        for (int mi = 0; mi < 2; ++mi) {
            const v16h a = load_a_frag(As, wm * 32 + mi * 16, lane);
            #pragma unroll
            for (int ni = 0; ni < 4; ++ni) {
                const v16h b = load_b_frag(Bs, wn * 64 + ni * 16, lane);
                acc[mi][ni] = __builtin_amdgcn_wmma_f32_16x16x32_f16(
                    false, a, false, b, (short)0, acc[mi][ni], false, false);
            }
        }
        __syncthreads();
    }

    // epilogue: bias + layerscale + NHWC->NCHW + residual
    #pragma unroll
    for (int mi = 0; mi < 2; ++mi) {
        const int mbase = mblk * 128 + wm * 32 + mi * 16 + (lane >> 4) * 8;
        #pragma unroll
        for (int ni = 0; ni < 4; ++ni) {
            const int c = nblk * 128 + wn * 64 + ni * 16 + (lane & 15);
            const float bn = bias[c];
            const float gm = gamma[c];
            #pragma unroll
            for (int g = 0; g < 8; ++g) {
                const int m  = mbase + g;            // NHWC position
                const int ww = m & 31, hh = (m >> 5) & 31, nb = m >> 10;
                const size_t oidx = (((size_t)(nb * DIM + c)) << 10) + (hh << 5) + ww;
                out[oidx] = x[oidx] + gm * (acc[mi][ni][g] + bn);
            }
        }
    }
}

// ---------------------------------------------------------------------------
extern "C" void kernel_launch(void* const* d_in, const int* in_sizes, int n_in,
                              void* d_out, int out_size, void* d_ws, size_t ws_size,
                              hipStream_t stream) {
    const float* x     = (const float*)d_in[0];
    const float* dw_w  = (const float*)d_in[1];
    const float* dw_b  = (const float*)d_in[2];
    const float* ln_w  = (const float*)d_in[3];
    const float* ln_b  = (const float*)d_in[4];
    const float* w1    = (const float*)d_in[5];
    const float* b1    = (const float*)d_in[6];
    const float* w2    = (const float*)d_in[7];
    const float* b2    = (const float*)d_in[8];
    const float* gamma = (const float*)d_in[9];
    float* out = (float*)d_out;

    char* ws = (char*)d_ws;
    _Float16* Aact = (_Float16*)(ws);                                  // 16 MB
    _Float16* W1h  = (_Float16*)(ws + (size_t)16 * 1024 * 1024);       //  2 MB
    _Float16* W2h  = (_Float16*)(ws + (size_t)18 * 1024 * 1024);       //  2 MB
    _Float16* Hid  = (_Float16*)(ws + (size_t)20 * 1024 * 1024);       // 64 MB

    dwconv_ln_kernel<<<NPOS, 256, 0, stream>>>(x, dw_w, dw_b, ln_w, ln_b, Aact);
    cvt_weights_kernel<<<(2 * DIM * HID) / 256, 256, 0, stream>>>(w1, w2, W1h, W2h);

    dim3 g1(HID / 128, NPOS / 128);   // (16, 128)
    gemm1_kernel<<<g1, 256, 0, stream>>>(Aact, W1h, b1, Hid);

    dim3 g2(DIM / 128, NPOS / 128);   // (4, 128)
    gemm2_kernel<<<g2, 256, 0, stream>>>(Hid, W2h, b2, gamma, x, out);
}
